// TransformerBlock_52140902974245
// MI455X (gfx1250) — matrix operand order repaired and checked by reading, not compiled, not run on hardware
//
#include <hip/hip_runtime.h>
#include <hip/hip_bf16.h>
#include <math.h>

// ---------------------------------------------------------------------------
// Transformer block for MI455X (gfx1250, wave32, WMMA + TDM).
// D=1024, H=16, HD=64, FF=4096, B=2, T=2048, M = B*T = 4096 tokens.
// ~137 GFLOP vs ~100MB traffic -> compute-bound -> v_wmma_f32_16x16x32_f16,
// with GEMM panels DMA'd into double-buffered LDS by the Tensor Data Mover
// (TDM prefetches tile k+1 while the matrix pipes consume tile k).
// ---------------------------------------------------------------------------

typedef __attribute__((ext_vector_type(16))) _Float16 v16h;
typedef __attribute__((ext_vector_type(8)))  float    v8f;

union ABFrag { v16h v; uint4 u4[2]; };
static_assert(sizeof(ABFrag) == 32, "frag is 8 VGPRs");

__device__ __forceinline__ v8f wmma_f16(const ABFrag& a, const ABFrag& b, v8f c) {
  return __builtin_amdgcn_wmma_f32_16x16x32_f16(false, a.v, false, b.v,
                                                (short)0, c, false, false);
}

__device__ __forceinline__ float gelu_tanh(float x) {
  const float c = 0.7978845608028654f; // sqrt(2/pi)
  float t = tanhf(c * (x + 0.044715f * x * x * x));
  return 0.5f * x * (1.0f + t);
}

// ---------------------------------------------------------------------------
// Tensor Data Mover: 2D f16 tile load Global->LDS (ISA ch. 8 descriptor).
// Padded LDS rows (pad_enable) stagger banks for conflict-free ds_load_b128.
// ---------------------------------------------------------------------------
#define HAS_TDM __has_builtin(__builtin_amdgcn_tensor_load_to_lds)

#if HAS_TDM
typedef __attribute__((ext_vector_type(4))) unsigned int tdm_u32x4;
typedef __attribute__((ext_vector_type(8))) int          tdm_i32x8;
typedef __attribute__((ext_vector_type(4))) int          tdm_i32x4;

__device__ __forceinline__ void tdm_load_2d_f16(
    unsigned int lds_byte_off, const void* gaddr,
    unsigned int tile_d0, unsigned int tile_d1, // elements/row, rows
    unsigned long long row_stride_elems,
    unsigned int pad_interval_enc, unsigned int pad_amount_enc) {
  unsigned long long ga = (unsigned long long)(uintptr_t)gaddr;
  tdm_u32x4 g0;
  g0[0] = 1u;                                     // count=1 (valid user D#)
  g0[1] = lds_byte_off;                           // LDS dest (bytes)
  g0[2] = (unsigned int)ga;                       // global_addr[31:0]
  g0[3] = (unsigned int)((ga >> 32) & 0x1FFFFFFu) // global_addr[56:32]
          | 0x80000000u;                          // type=2 ("image")
  tdm_i32x8 g1;
  g1[0] = (int)((1u << 16)                        // data_size=1 -> 2 bytes
                | (1u << 20)                      // pad_enable
                | (pad_interval_enc << 22) | (pad_amount_enc << 25));
  g1[1] = (int)((tile_d0 & 0xFFFFu) << 16);       // tensor_dim0 = tile_d0
  g1[2] = (int)((tile_d0 >> 16) | ((tile_d1 & 0xFFFFu) << 16)); // tensor_dim1
  g1[3] = (int)((tile_d1 >> 16) | (tile_d0 << 16));             // tile_dim0
  g1[4] = (int)tile_d1;                           // tile_dim1 (tile_dim2=0)
  g1[5] = (int)(unsigned int)(row_stride_elems & 0xFFFFFFFFu);  // dim0_stride lo
  g1[6] = (int)(unsigned int)((row_stride_elems >> 32) & 0xFFFFu);
  g1[7] = 0;
  tdm_i32x4 z4 = {0, 0, 0, 0};
#if defined(__clang_major__) && __clang_major__ >= 23
  tdm_i32x8 z8 = {0, 0, 0, 0, 0, 0, 0, 0};
  __builtin_amdgcn_tensor_load_to_lds(g0, g1, z4, z4, z8, 0);
#else
  __builtin_amdgcn_tensor_load_to_lds(g0, g1, z4, z4, 0);
#endif
}
#endif // HAS_TDM

// ---------------------------------------------------------------------------
// Weight convert + transpose: W fp32 [N,K] row-major -> Wt f16 [K,N].
// ---------------------------------------------------------------------------
__global__ void cvtT_kernel(const float* __restrict__ w, _Float16* __restrict__ wt,
                            int N, int K) {
  size_t idx = (size_t)blockIdx.x * blockDim.x + threadIdx.x;
  int    n   = (int)(idx % N);
  size_t k   = idx / N;
  wt[k * N + n] = (_Float16)w[(size_t)n * K + k];
}

// ---------------------------------------------------------------------------
// LayerNorm fp32 [row,1024] -> f16, one row per 256-thread block (8 waves).
// ---------------------------------------------------------------------------
__global__ void ln_f16_kernel(const float* __restrict__ x, const float* __restrict__ g,
                              const float* __restrict__ s, _Float16* __restrict__ out) {
  const int Dm = 1024;
  int row = blockIdx.x;
  float4 v = ((const float4*)(x + (size_t)row * Dm))[threadIdx.x];
  float sum = v.x + v.y + v.z + v.w;
  float sq  = v.x * v.x + v.y * v.y + v.z * v.z + v.w * v.w;
#pragma unroll
  for (int o = 16; o > 0; o >>= 1) {
    sum += __shfl_xor(sum, o, 32);
    sq  += __shfl_xor(sq,  o, 32);
  }
  __shared__ float ssum[8], ssq[8];
  int wid = threadIdx.x >> 5;
  if ((threadIdx.x & 31) == 0) { ssum[wid] = sum; ssq[wid] = sq; }
  __syncthreads();
  float ts = 0.f, tq = 0.f;
#pragma unroll
  for (int i = 0; i < 8; ++i) { ts += ssum[i]; tq += ssq[i]; }
  float mean = ts * (1.0f / Dm);
  float inv  = rsqrtf(tq * (1.0f / Dm) - mean * mean + 1e-5f);
  int c0 = threadIdx.x * 4;
  _Float16* orow = out + (size_t)row * Dm;
  float xv[4] = {v.x, v.y, v.z, v.w};
#pragma unroll
  for (int i = 0; i < 4; ++i)
    orow[c0 + i] = (_Float16)(g[c0 + i] * (xv[i] - mean) * inv + s[c0 + i]);
}

// ---------------------------------------------------------------------------
// Double-buffered, TDM-fed WMMA GEMM: C[M,N] = A[M,K](f16) @ Bt[K,N](f16).
// Block = 128 threads = 4 waves; block tile 128x128; wave tile 64x64;
// K-step 64 (two 32-wide WMMA chunks). Pipeline per step:
//   wave0 issues TDM for step k+1 -> alternate buffer
//   all waves: 32 WMMAs from current buffer (ds_load_b128 frags)
//   wave0: s_wait_tensorcnt 0 ; all: one barrier
// A row pitch 72 halves (144B), B row pitch 136 halves (272B): bank-staggered
// via TDM pad_enable. 2 x (18KB + 17KB) = 70KB LDS of the WGP's 320KB.
// MODE 1: +bias, gelu -> f16   MODE 2: +bias,+residual -> f32
// MODE 3: scatter f16 [B,H,T,HD]   MODE 4: scatter f16 [B,H,HD,T]
// ---------------------------------------------------------------------------
#define APITCH 72   // 64 halves + 4 DWORDs pad
#define BPITCH 136  // 128 halves + 4 DWORDs pad

__device__ __forceinline__ void stage_tiles(const _Float16* A, const _Float16* Bt,
                                            int N, int K, int m0b, int n0b, int k0,
                                            _Float16* bufA, _Float16* bufB,
                                            int wave, int tid) {
#if HAS_TDM
  (void)tid;
  if (wave == 0) {
    // A tile: 64 elems/row x 128 rows; pad every 32 DWORDs (+4 DWORDs)
    tdm_load_2d_f16((unsigned int)(uintptr_t)bufA,
                    A + (size_t)m0b * K + k0, 64u, 128u,
                    (unsigned long long)K, 4u, 3u);
    // B tile: 128 elems/row x 64 rows; pad every 64 DWORDs (+4 DWORDs)
    tdm_load_2d_f16((unsigned int)(uintptr_t)bufB,
                    Bt + (size_t)k0 * N + n0b, 128u, 64u,
                    (unsigned long long)N, 5u, 3u);
  }
#else
  (void)wave;
  { // A: thread t copies row t (64 halves = 8x uint4)
    const uint4* src = (const uint4*)(A + (size_t)(m0b + tid) * K + k0);
    uint4* dst = (uint4*)(bufA + tid * APITCH);
#pragma unroll
    for (int q = 0; q < 8; ++q) dst[q] = src[q];
  }
  { // B: thread t copies half (t&1) of row (t>>1)
    int r = tid >> 1, hf = tid & 1;
    const uint4* src = (const uint4*)(Bt + (size_t)(k0 + r) * N + n0b + hf * 64);
    uint4* dst = (uint4*)(bufB + r * BPITCH + hf * 64);
#pragma unroll
    for (int q = 0; q < 8; ++q) dst[q] = src[q];
  }
#endif
}

__device__ __forceinline__ void stage_fence(int wave) {
#if HAS_TDM
  if (wave == 0) __builtin_amdgcn_s_wait_tensorcnt(0);
#else
  (void)wave; // ds_store/loadcnt waits inserted by compiler before the barrier
#endif
}

template <int MODE>
__global__ void __launch_bounds__(128)
gemm_kernel(const _Float16* __restrict__ A, const _Float16* __restrict__ Bt,
            const float* __restrict__ bias, const float* __restrict__ res,
            void* __restrict__ outp, int M, int N, int K) {
  const int tid  = threadIdx.x;
  const int lane = tid & 31;
  const int wave = tid >> 5;
  const int m0b = blockIdx.y * 128;
  const int n0b = blockIdx.x * 128;
  const int wm = (wave >> 1) * 64;
  const int wn = (wave & 1) * 64;
  const int am = lane & 15;        // A frag: M row
  const int ak = (lane >> 4) * 8;  // A frag: K sub-offset

  __shared__ __align__(16) _Float16 shA[2][128 * APITCH];
  __shared__ __align__(16) _Float16 shB[2][64 * BPITCH];

  // prologue: stage k0 = 0 into buffer 0
  stage_tiles(A, Bt, N, K, m0b, n0b, 0, shA[0], shB[0], wave, tid);
  stage_fence(wave);
  __syncthreads();

  v8f acc[4][4] = {};
  for (int k0 = 0; k0 < K; k0 += 64) {
    const int cur = (k0 >> 6) & 1;
    if (k0 + 64 < K) // async prefetch of next K panel into alternate buffer
      stage_tiles(A, Bt, N, K, m0b, n0b, k0 + 64, shA[cur ^ 1], shB[cur ^ 1],
                  wave, tid);

#pragma unroll
    for (int c = 0; c < 2; ++c) { // two 32-wide WMMA chunks
      ABFrag a[4], b[4];
#pragma unroll
      for (int i = 0; i < 4; ++i) { // A 16x32 frag: lane=m, {k..k+7, k+16..k+23}
        const _Float16* p = shA[cur] + (wm + 16 * i + am) * APITCH + 32 * c + ak;
        a[i].u4[0] = *(const uint4*)p;
        a[i].u4[1] = *(const uint4*)(p + 16);
      }
#pragma unroll
      for (int j = 0; j < 4; ++j) { // B 32x16 frag: lane=n, k in A's order
        const _Float16* p = shB[cur] + (32 * c) * BPITCH + wn + 16 * j + am;
#pragma unroll
        for (int s = 0; s < 16; ++s)
          b[j].v[s] = p[((s & 7) + ak + (s & 8) * 2) * BPITCH];
      }
#pragma unroll
      for (int i = 0; i < 4; ++i)
#pragma unroll
        for (int j = 0; j < 4; ++j)
          acc[i][j] = wmma_f16(a[i], b[j], acc[i][j]);
    }

    stage_fence(wave);  // next buffer complete (TDM) before publishing
    __syncthreads();    // one barrier per K-step
  }

  // C layout: VGPR r, lane l -> (Mrel = r + 8*(l/16), Nrel = l%16)
  const int chalf = (lane >> 4) * 8;
  const int cn = lane & 15;
#pragma unroll
  for (int i = 0; i < 4; ++i)
#pragma unroll
    for (int j = 0; j < 4; ++j)
#pragma unroll
      for (int r = 0; r < 8; ++r) {
        int row = m0b + wm + 16 * i + r + chalf;
        int col = n0b + wn + 16 * j + cn;
        float v = acc[i][j][r];
        if (MODE == 1) {
          ((_Float16*)outp)[(size_t)row * N + col] = (_Float16)gelu_tanh(v + bias[col]);
        } else if (MODE == 2) {
          ((float*)outp)[(size_t)row * N + col] =
              v + bias[col] + res[(size_t)row * N + col];
        } else if (MODE == 3) { // [B,H,T,HD], T=2048, HD=64, H=16
          int b_ = row >> 11, t = row & 2047, h = col >> 6, hd = col & 63;
          ((_Float16*)outp)[((((size_t)b_ * 16 + h) * 2048 + t) << 6) + hd] = (_Float16)v;
        } else if (MODE == 4) { // [B,H,HD,T]
          int b_ = row >> 11, t = row & 2047, h = col >> 6, hd = col & 63;
          ((_Float16*)outp)[((((size_t)b_ * 16 + h) * 64 + hd) << 11) + t] = (_Float16)v;
        }
      }
}

// ---------------------------------------------------------------------------
// Causal flash attention, one (b,h) x 64-query tile per block (4 waves,
// 16 queries/wave). Scores and P@V via WMMA; online softmax row stats reduced
// across the 16-lane half that owns each C-layout row; P re-laid-out via LDS.
// K/V panels live in the 192MB L2; B-frag loads are 16-bit gathers strided by T (K) or HD (V).
// ---------------------------------------------------------------------------
__global__ void __launch_bounds__(128)
attn_kernel(const _Float16* __restrict__ Q, const _Float16* __restrict__ KT,
            const _Float16* __restrict__ V, _Float16* __restrict__ ctx) {
  const int T = 2048, HD = 64;
  const int bh   = blockIdx.y;
  const int lane = threadIdx.x & 31;
  const int wave = threadIdx.x >> 5;
  const int qb   = blockIdx.x * 64 + wave * 16;
  __shared__ __align__(16) _Float16 pshm[4][16 * 32];

  const _Float16* Qb = Q  + (size_t)bh * T * HD;
  const _Float16* Kb = KT + (size_t)bh * HD * T;
  const _Float16* Vb = V  + (size_t)bh * T * HD;

  const int am = lane & 15;
  const int ak = (lane >> 4) * 8;
  const int qhalf = (lane >> 4) * 8;

  ABFrag aq[2]; // Q 16x64 = two 16x32 A-frags
#pragma unroll
  for (int c = 0; c < 2; ++c) {
    const _Float16* p = Qb + (size_t)(qb + am) * HD + 32 * c + ak;
    aq[c].u4[0] = *(const uint4*)p;
    aq[c].u4[1] = *(const uint4*)(p + 16);
  }

  v8f acc[4] = {};
  float mrow[8], lrow[8];
#pragma unroll
  for (int r = 0; r < 8; ++r) { mrow[r] = -INFINITY; lrow[r] = 0.f; }

  for (int kt = 0; kt < qb + 16; kt += 32) { // causal key range
    v8f s[2];
#pragma unroll
    for (int jj = 0; jj < 2; ++jj) {
      ABFrag bk0, bk1; // from KT: lane=key column, elements over hd in A's k order
      const _Float16* p0 = Kb + kt + 16 * jj + am;
      const _Float16* p1 = Kb + (size_t)32 * T + kt + 16 * jj + am;
#pragma unroll
      for (int s = 0; s < 16; ++s) {
        int kk = (s & 7) + ak + (s & 8) * 2;
        bk0.v[s] = p0[(size_t)kk * T]; bk1.v[s] = p1[(size_t)kk * T];
      }
      v8f z = {};
      s[jj] = wmma_f16(aq[1], bk1, wmma_f16(aq[0], bk0, z));
      int n = kt + 16 * jj + am;
#pragma unroll
      for (int r = 0; r < 8; ++r) {
        int m = qb + r + qhalf;
        s[jj][r] = (n <= m) ? s[jj][r] * 0.125f : -INFINITY; // 1/sqrt(64)
      }
    }
#pragma unroll
    for (int r = 0; r < 8; ++r) { // online softmax per query row
      float mx = fmaxf(s[0][r], s[1][r]);
#pragma unroll
      for (int o = 1; o < 16; o <<= 1) mx = fmaxf(mx, __shfl_xor(mx, o, 32));
      float mnew = fmaxf(mrow[r], mx);
      float corr = __expf(mrow[r] - mnew);
      float p0 = __expf(s[0][r] - mnew);
      float p1 = __expf(s[1][r] - mnew);
      float rs = p0 + p1;
#pragma unroll
      for (int o = 1; o < 16; o <<= 1) rs += __shfl_xor(rs, o, 32);
      lrow[r] = lrow[r] * corr + rs;
      mrow[r] = mnew;
#pragma unroll
      for (int j = 0; j < 4; ++j) acc[j][r] *= corr;
      int mloc = r + qhalf; // stage P (wave-private; DS in-order per wave)
      pshm[wave][mloc * 32 + am]      = (_Float16)p0;
      pshm[wave][mloc * 32 + 16 + am] = (_Float16)p1;
    }
    ABFrag ap; // reload P in A-layout
    const _Float16* pp = &pshm[wave][am * 32 + ak];
    ap.u4[0] = *(const uint4*)pp;
    ap.u4[1] = *(const uint4*)(pp + 16);
#pragma unroll
    for (int j = 0; j < 4; ++j) { // V B-frags: lane=hd, elements over keys in A's k order
      ABFrag bv;
      const _Float16* vp = Vb + (size_t)kt * HD + 16 * j + am;
#pragma unroll
      for (int s = 0; s < 16; ++s)
        bv.v[s] = vp[((s & 7) + ak + (s & 8) * 2) * HD];
      acc[j] = wmma_f16(ap, bv, acc[j]);
    }
  }

  const int b_ = bh >> 4, h = bh & 15;
#pragma unroll
  for (int r = 0; r < 8; ++r) { // ctx -> token-major [B*T, 1024] f16
    float rec = 1.0f / lrow[r];
    int m = qb + r + qhalf;
    size_t base = ((size_t)b_ * T + m) * 1024 + (size_t)h * 64;
#pragma unroll
    for (int j = 0; j < 4; ++j)
      ctx[base + 16 * j + am] = (_Float16)(acc[j][r] * rec);
  }
}

// ---------------------------------------------------------------------------
// Host orchestration.
// ---------------------------------------------------------------------------
extern "C" void kernel_launch(void* const* d_in, const int* in_sizes, int n_in,
                              void* d_out, int out_size, void* d_ws, size_t ws_size,
                              hipStream_t stream) {
  (void)in_sizes; (void)n_in; (void)out_size; (void)ws_size;
  const float* x  = (const float*)d_in[0];
  const float* wq = (const float*)d_in[1];
  const float* wk = (const float*)d_in[2];
  const float* wv = (const float*)d_in[3];
  const float* wo = (const float*)d_in[4];
  const float* bo = (const float*)d_in[5];
  const float* w1 = (const float*)d_in[6];
  const float* b1 = (const float*)d_in[7];
  const float* w2 = (const float*)d_in[8];
  const float* b2 = (const float*)d_in[9];
  const float* g1 = (const float*)d_in[10];
  const float* s1 = (const float*)d_in[11];
  const float* g2 = (const float*)d_in[12];
  const float* s2 = (const float*)d_in[13];

  const int D = 1024, FF = 4096, M = 4096;

  char* ws = (char*)d_ws;
  size_t off = 0;
  auto carve = [&](size_t bytes) -> char* {
    char* p = ws + off;
    off += (bytes + 255) & ~(size_t)255;
    return p;
  };
  _Float16* WQT = (_Float16*)carve((size_t)D * D * 2);
  _Float16* WKT = (_Float16*)carve((size_t)D * D * 2);
  _Float16* WVT = (_Float16*)carve((size_t)D * D * 2);
  _Float16* WOT = (_Float16*)carve((size_t)D * D * 2);
  _Float16* W1T = (_Float16*)carve((size_t)D * FF * 2);
  _Float16* W2T = (_Float16*)carve((size_t)FF * D * 2);
  _Float16* H16 = (_Float16*)carve((size_t)M * D * 2);   // LN output (reused)
  _Float16* Qb  = (_Float16*)carve((size_t)M * D * 2);   // [B,H,T,HD]
  _Float16* KTb = (_Float16*)carve((size_t)M * D * 2);   // [B,H,HD,T]
  _Float16* Vb  = (_Float16*)carve((size_t)M * D * 2);   // [B,H,T,HD]
  _Float16* CTX = (_Float16*)carve((size_t)M * D * 2);   // token-major
  float*    XMD = (float*)   carve((size_t)M * D * 4);   // x + attn out (fp32)
  _Float16* MID = (_Float16*)carve((size_t)M * FF * 2);  // gelu(h2@w1^T+b1)

  cvtT_kernel<<<(D * D) / 256, 256, 0, stream>>>(wq, WQT, D, D);
  cvtT_kernel<<<(D * D) / 256, 256, 0, stream>>>(wk, WKT, D, D);
  cvtT_kernel<<<(D * D) / 256, 256, 0, stream>>>(wv, WVT, D, D);
  cvtT_kernel<<<(D * D) / 256, 256, 0, stream>>>(wo, WOT, D, D);
  cvtT_kernel<<<(FF * D) / 256, 256, 0, stream>>>(w1, W1T, FF, D);
  cvtT_kernel<<<(D * FF) / 256, 256, 0, stream>>>(w2, W2T, D, FF);

  // pre-LN attention
  ln_f16_kernel<<<M, 256, 0, stream>>>(x, g1, s1, H16);
  dim3 gDD(D / 128, M / 128);
  gemm_kernel<3><<<gDD, 128, 0, stream>>>(H16, WQT, nullptr, nullptr, Qb,  M, D, D);
  gemm_kernel<4><<<gDD, 128, 0, stream>>>(H16, WKT, nullptr, nullptr, KTb, M, D, D);
  gemm_kernel<3><<<gDD, 128, 0, stream>>>(H16, WVT, nullptr, nullptr, Vb,  M, D, D);
  attn_kernel<<<dim3(2048 / 64, 32), 128, 0, stream>>>(Qb, KTb, Vb, CTX);
  gemm_kernel<2><<<gDD, 128, 0, stream>>>(CTX, WOT, bo, x, XMD, M, D, D);

  // pre-LN MLP
  ln_f16_kernel<<<M, 256, 0, stream>>>(XMD, g2, s2, H16);
  gemm_kernel<1><<<dim3(FF / 128, M / 128), 128, 0, stream>>>(H16, W1T, b1, nullptr, MID, M, FF, D);
  gemm_kernel<2><<<dim3(D / 128, M / 128), 128, 0, stream>>>(MID, W2T, b2, XMD, (float*)d_out, M, D, FF);
}
